// PointNetSetAbstraction_3427383902450
// MI455X (gfx1250) — matrix operand order repaired and checked by reading, not compiled, not run on hardware
//
#include <hip/hip_runtime.h>
#include <hip/hip_bf16.h>

typedef __attribute__((ext_vector_type(16))) _Float16 v16h;
typedef __attribute__((ext_vector_type(8)))  float    v8f;

#define BATCH   16
#define NPTS    4096
#define NPOINT  1024
#define NSAMPLE 32
#define DIN     64
#define RADIUS2 0.01f
#define EPS_BN  1e-5f
#define MTOT    (BATCH * NPOINT * NSAMPLE)   // 524288 positions

// ---------------------------------------------------------------- utilities
__global__ void zero_f32_kernel(float* __restrict__ p, int n) {
    int i = blockIdx.x * blockDim.x + threadIdx.x;
    if (i < n) p[i] = 0.0f;
}

// f32 weights -> f16, padded along Cin to CinPad with zeros.  Wh[o][c]
__global__ void convert_w_kernel(const float* __restrict__ W, _Float16* __restrict__ Wh,
                                 int Cout, int Cin, int CinPad) {
    int i = blockIdx.x * blockDim.x + threadIdx.x;
    if (i >= Cout * CinPad) return;
    int c = i % CinPad, o = i / CinPad;
    Wh[i] = (c < Cin) ? (_Float16)W[o * Cin + c] : (_Float16)0.0f;
}

// ---------------------------------------------------------------- 1) FPS
// One block per batch. xyz staged in LDS, running distances in registers,
// shfl + LDS two-level argmax per step. Writes new_xyz (B,3,S) into d_out.
__global__ void __launch_bounds__(1024) fps_kernel(const float* __restrict__ xyz,
                                                   int* __restrict__ fpsIdx,
                                                   float* __restrict__ newXyzOut) {
    __shared__ float sx[NPTS], sy[NPTS], sz[NPTS];
    __shared__ float rv[32];
    __shared__ int   ri[32];
    __shared__ int   sFar;

    const int b = blockIdx.x, tid = threadIdx.x;
    const float* xb = xyz + (size_t)b * 3 * NPTS;
    for (int n = tid; n < NPTS; n += 1024) {
        sx[n] = xb[n];
        sy[n] = xb[NPTS + n];
        sz[n] = xb[2 * NPTS + n];
    }
    float dist[4];
    int   pid[4];
#pragma unroll
    for (int j = 0; j < 4; ++j) { pid[j] = tid + j * 1024; dist[j] = 1e10f; }
    if (tid == 0) sFar = 0;
    __syncthreads();

    const int lane = tid & 31, wave = tid >> 5;
    for (int it = 0; it < NPOINT; ++it) {
        const int far = sFar;
        if (tid == 0) {
            fpsIdx[b * NPOINT + it] = far;
            newXyzOut[(size_t)b * 3 * NPOINT + it]              = sx[far];
            newXyzOut[(size_t)b * 3 * NPOINT + NPOINT + it]     = sy[far];
            newXyzOut[(size_t)b * 3 * NPOINT + 2 * NPOINT + it] = sz[far];
        }
        const float cx = sx[far], cy = sy[far], cz = sz[far];
        float bestV = -1.0f; int bestI = 0;
#pragma unroll
        for (int j = 0; j < 4; ++j) {
            float dx = sx[pid[j]] - cx, dy = sy[pid[j]] - cy, dz = sz[pid[j]] - cz;
            float d2 = dx * dx + dy * dy + dz * dz;
            dist[j] = fminf(dist[j], d2);
            if (dist[j] > bestV || (dist[j] == bestV && pid[j] < bestI)) {
                bestV = dist[j]; bestI = pid[j];
            }
        }
#pragma unroll
        for (int off = 16; off > 0; off >>= 1) {
            float ov = __shfl_xor(bestV, off, 32);
            int   oi = __shfl_xor(bestI, off, 32);
            if (ov > bestV || (ov == bestV && oi < bestI)) { bestV = ov; bestI = oi; }
        }
        if (lane == 0) { rv[wave] = bestV; ri[wave] = bestI; }
        __syncthreads();
        if (wave == 0) {
            bestV = rv[lane]; bestI = ri[lane];
#pragma unroll
            for (int off = 16; off > 0; off >>= 1) {
                float ov = __shfl_xor(bestV, off, 32);
                int   oi = __shfl_xor(bestI, off, 32);
                if (ov > bestV || (ov == bestV && oi < bestI)) { bestV = ov; bestI = oi; }
            }
            if (lane == 0) sFar = bestI;
        }
        __syncthreads();
    }
}

// ---------------------------------------------------------------- 2) ball query
// One thread per query; ordered scan == sort-then-take-first-32 of reference.
__global__ void __launch_bounds__(256) ball_query_kernel(const float* __restrict__ xyz,
                                                         const float* __restrict__ newXyz,
                                                         int* __restrict__ grpIdx) {
    __shared__ float sx[NPTS], sy[NPTS], sz[NPTS];
    const int b = blockIdx.y;
    const int s = blockIdx.x * blockDim.x + threadIdx.x;
    const float* xb = xyz + (size_t)b * 3 * NPTS;
    for (int n = threadIdx.x; n < NPTS; n += 256) {
        sx[n] = xb[n]; sy[n] = xb[NPTS + n]; sz[n] = xb[2 * NPTS + n];
    }
    __syncthreads();
    const float qx = newXyz[(size_t)b * 3 * NPOINT + s];
    const float qy = newXyz[(size_t)b * 3 * NPOINT + NPOINT + s];
    const float qz = newXyz[(size_t)b * 3 * NPOINT + 2 * NPOINT + s];
    int* out = grpIdx + ((size_t)b * NPOINT + s) * NSAMPLE;
    int cnt = 0, first = 0;
    for (int n = 0; n < NPTS && cnt < NSAMPLE; ++n) {
        float dx = sx[n] - qx, dy = sy[n] - qy, dz = sz[n] - qz;
        if (dx * dx + dy * dy + dz * dz <= RADIUS2) {
            if (cnt == 0) first = n;
            out[cnt++] = n;
        }
    }
    for (; cnt < NSAMPLE; ++cnt) out[cnt] = first;
}

// ---------------------------------------------------------------- 3) gather -> X0 f16 [m][96]
__global__ void gather_kernel(const float* __restrict__ xyz, const float* __restrict__ points,
                              const float* __restrict__ newXyz, const int* __restrict__ grpIdx,
                              _Float16* __restrict__ X0) {
    const int m = blockIdx.x * blockDim.x + threadIdx.x;      // m = ((b*S+s)*K + k)
    const int s = (m >> 5) & (NPOINT - 1);
    const int b = m >> 15;
    const int n = grpIdx[m];
    const float* xb = xyz + (size_t)b * 3 * NPTS;
    _Float16* row = X0 + (size_t)m * 96;
    row[0] = (_Float16)(xb[n]            - newXyz[(size_t)b * 3 * NPOINT + s]);
    row[1] = (_Float16)(xb[NPTS + n]     - newXyz[(size_t)b * 3 * NPOINT + NPOINT + s]);
    row[2] = (_Float16)(xb[2 * NPTS + n] - newXyz[(size_t)b * 3 * NPOINT + 2 * NPOINT + s]);
    const float* pb = points + (size_t)b * DIN * NPTS;
#pragma unroll 8
    for (int c = 0; c < DIN; ++c) row[3 + c] = (_Float16)pb[c * NPTS + n];
#pragma unroll
    for (int c = 3 + DIN; c < 96; ++c) row[c] = (_Float16)0.0f;
}

// ---------------------------------------------------------------- 4) WMMA GEMM + bias + BN stats
// D[co][m] = sum_c W[co][c]*X[m][c] + bias.  Wave = one 16x16 tile, 8 waves/block.
// CINPAD/COUT compile-time -> fully unrolled straight-line K-loop (2 or 3 WMMAs).
// Fuses per-channel sum / sum-of-squares via shfl -> LDS -> global atomics.
template <int CINPAD, int COUT>
__global__ void __launch_bounds__(256) gemm_wmma_kernel(const _Float16* __restrict__ X,
                                                        const _Float16* __restrict__ Wh,
                                                        const float* __restrict__ bias,
                                                        float* __restrict__ Y,
                                                        float* __restrict__ stats) {
    __shared__ float ls[32];
    const int tid = threadIdx.x;
    if (tid < 32) ls[tid] = 0.0f;
    __syncthreads();

    const int lane = tid & 31, wave = tid >> 5;
    const int col = lane & 15, half = lane >> 4;
    const int r0 = blockIdx.y << 4;
    const int m0 = ((blockIdx.x << 3) + wave) << 4;

    v8f acc = {};
    const _Float16* wrow = Wh + (size_t)(r0 + col) * CINPAD;                 // A row = lane&15
    const _Float16* xrow = X + (size_t)(m0 + col) * CINPAD + (half << 4);    // B col = lane&15

#pragma unroll
    for (int k0 = 0; k0 < CINPAD; k0 += 32) {
        v16h a;
#pragma unroll
        for (int e = 0; e < 16; ++e)   // contiguous K order (k = 16*half + e), matching B
            a[e] = wrow[k0 + (half << 4) + e];
        v16h bf = *(const v16h*)(xrow + k0);   // 16 contiguous K per lane
        acc = __builtin_amdgcn_wmma_f32_16x16x32_f16(false, a, false, bf,
                                                     (short)0, acc, false, false);
    }

    const int coBase = r0 + (half << 3);
    float* yp = Y + (size_t)(m0 + col) * COUT + coBase;
#pragma unroll
    for (int j = 0; j < 8; ++j) {
        float v = acc[j] + bias[coBase + j];
        yp[j] = v;
        float s = v, q = v * v;
#pragma unroll
        for (int off = 1; off < 16; off <<= 1) {   // reduce across 16 columns (half-wave)
            s += __shfl_xor(s, off, 32);
            q += __shfl_xor(q, off, 32);
        }
        if (col == 0) {
            atomicAdd(&ls[(half << 3) + j], s);
            atomicAdd(&ls[16 + (half << 3) + j], q);
        }
    }
    __syncthreads();
    if (tid < 16) {
        atomicAdd(&stats[r0 + tid], ls[tid]);
        atomicAdd(&stats[COUT + r0 + tid], ls[16 + tid]);
    }
}

// ---------------------------------------------------------------- 5) BN + ReLU -> f16 next input
__global__ void bn_relu_kernel(const float* __restrict__ Y, const float* __restrict__ stats,
                               const float* __restrict__ gamma, const float* __restrict__ beta,
                               _Float16* __restrict__ Xn, int Cout) {
    size_t i = (size_t)blockIdx.x * blockDim.x + threadIdx.x;
    int co = (int)(i % Cout);
    float mean = stats[co] * (1.0f / (float)MTOT);
    float var  = stats[Cout + co] * (1.0f / (float)MTOT) - mean * mean;
    float sc = gamma[co] * rsqrtf(var + EPS_BN);
    float sh = beta[co] - mean * sc;
    Xn[i] = (_Float16)fmaxf(Y[i] * sc + sh, 0.0f);
}

// ---------------------------------------------------------------- 6) BN + ReLU + max over K
__global__ void bn_max_kernel(const float* __restrict__ Y, const float* __restrict__ stats,
                              const float* __restrict__ gamma, const float* __restrict__ beta,
                              float* __restrict__ out) {
    const int i = blockIdx.x * blockDim.x + threadIdx.x;   // (b, s, co)
    const int co = i & 127;
    const int s = (i >> 7) & (NPOINT - 1);
    const int b = i >> 17;
    float mean = stats[co] * (1.0f / (float)MTOT);
    float var  = stats[128 + co] * (1.0f / (float)MTOT) - mean * mean;
    float sc = gamma[co] * rsqrtf(var + EPS_BN);
    float sh = beta[co] - mean * sc;
    const float* yp = Y + (size_t)((b * NPOINT + s) * NSAMPLE) * 128 + co;
    float mx = -3.4e38f;
#pragma unroll
    for (int k = 0; k < NSAMPLE; ++k)
        mx = fmaxf(mx, fmaxf(yp[k * 128] * sc + sh, 0.0f));
    out[((size_t)b * 128 + co) * NPOINT + s] = mx;
}

// ---------------------------------------------------------------- launch
extern "C" void kernel_launch(void* const* d_in, const int* in_sizes, int n_in,
                              void* d_out, int out_size, void* d_ws, size_t ws_size,
                              hipStream_t stream) {
    const float* xyz    = (const float*)d_in[0];
    const float* points = (const float*)d_in[1];
    const float* W0 = (const float*)d_in[2];  const float* b0 = (const float*)d_in[3];
    const float* g0 = (const float*)d_in[4];  const float* be0 = (const float*)d_in[5];
    const float* W1 = (const float*)d_in[6];  const float* b1 = (const float*)d_in[7];
    const float* g1 = (const float*)d_in[8];  const float* be1 = (const float*)d_in[9];
    const float* W2 = (const float*)d_in[10]; const float* b2 = (const float*)d_in[11];
    const float* g2 = (const float*)d_in[12]; const float* be2 = (const float*)d_in[13];

    char* p = (char*)d_ws;
    auto carve = [&](size_t bytes) { void* r = (void*)p; p += (bytes + 255) & ~(size_t)255; return r; };
    int*      fpsIdx = (int*)carve((size_t)BATCH * NPOINT * 4);
    int*      grpIdx = (int*)carve((size_t)BATCH * NPOINT * NSAMPLE * 4);
    _Float16* Wh0    = (_Float16*)carve(64 * 96 * 2);
    _Float16* Wh1    = (_Float16*)carve(64 * 64 * 2);
    _Float16* Wh2    = (_Float16*)carve(128 * 64 * 2);
    float*    st0    = (float*)carve(256 * 4);
    float*    st1    = (float*)carve(256 * 4);
    float*    st2    = (float*)carve(256 * 4);
    _Float16* Xb     = (_Float16*)carve((size_t)MTOT * 96 * 2);
    float*    Yb     = (float*)carve((size_t)MTOT * 128 * 4);

    float* outNewXyz = (float*)d_out;                        // (B,3,S)
    float* outPoints = (float*)d_out + (size_t)BATCH * 3 * NPOINT;  // (B,128,S)

    zero_f32_kernel<<<3, 256, 0, stream>>>(st0, 768);        // st0..st2 contiguous

    convert_w_kernel<<<(64 * 96 + 255) / 256, 256, 0, stream>>>(W0, Wh0, 64, 67, 96);
    convert_w_kernel<<<(64 * 64 + 255) / 256, 256, 0, stream>>>(W1, Wh1, 64, 64, 64);
    convert_w_kernel<<<(128 * 64 + 255) / 256, 256, 0, stream>>>(W2, Wh2, 128, 64, 64);

    fps_kernel<<<BATCH, 1024, 0, stream>>>(xyz, fpsIdx, outNewXyz);
    ball_query_kernel<<<dim3(NPOINT / 256, BATCH), 256, 0, stream>>>(xyz, outNewXyz, grpIdx);
    gather_kernel<<<MTOT / 256, 256, 0, stream>>>(xyz, points, outNewXyz, grpIdx, Xb);

    // layer 0: 96(pad of 67) -> 64
    gemm_wmma_kernel<96, 64><<<dim3(MTOT / 128, 4), 256, 0, stream>>>(Xb, Wh0, b0, Yb, st0);
    bn_relu_kernel<<<(int)(((size_t)MTOT * 64) / 256), 256, 0, stream>>>(Yb, st0, g0, be0, Xb, 64);
    // layer 1: 64 -> 64
    gemm_wmma_kernel<64, 64><<<dim3(MTOT / 128, 4), 256, 0, stream>>>(Xb, Wh1, b1, Yb, st1);
    bn_relu_kernel<<<(int)(((size_t)MTOT * 64) / 256), 256, 0, stream>>>(Yb, st1, g1, be1, Xb, 64);
    // layer 2: 64 -> 128, then BN + ReLU + max over K fused into output
    gemm_wmma_kernel<64, 128><<<dim3(MTOT / 128, 8), 256, 0, stream>>>(Xb, Wh2, b2, Yb, st2);
    bn_max_kernel<<<(BATCH * NPOINT * 128) / 256, 256, 0, stream>>>(Yb, st2, g2, be2, outPoints);
}
